// NetEnsemble_54803782697253
// MI455X (gfx1250) — matrix operand order repaired and checked by reading, not compiled, not run on hardware
//
#include <hip/hip_runtime.h>

// ---------------------------------------------------------------------------
// MI455X (gfx1250) denoiser pipeline: STFT -> gating GRU -> routing ->
// grouped specialist GRU -> decoder mask -> iSTFT.
// All GEMM-shaped work (input projections, recurrent h*Whh^T, decoder) runs
// on v_wmma_f32_16x16x32_bf16. GEMM waves are register-tiled 2x2 fragments
// to double WMMA-per-byte; recurrent scans keep h in LDS and batch all
// samples (or per-expert groups) into single workgroups.
// ---------------------------------------------------------------------------

#define FFTN  1024
#define HOPSZ 256
#define NFR   626      // 1 + (160000+1024-1024)/256
#define MPAD  640      // NFR padded for WMMA M tiles (also /128 block tiles)
#define FBIN  513
#define FPAD  544      // 513 padded to x32 for WMMA K
#define NB    32
#define HID   256
#define G3    768      // 3*HID
#define TLEN  160000
#define NEXP  10
#define DECNP 576      // 513 padded to x64 (block N-tile) for decoder weights
#define TWOPI 6.28318530717958647692f

typedef __attribute__((ext_vector_type(16))) __bf16         v16bf;
typedef __attribute__((ext_vector_type(8)))  float          v8f;
typedef __attribute__((ext_vector_type(16))) unsigned short v16u;
typedef __attribute__((ext_vector_type(8)))  unsigned short v8u;

__device__ __forceinline__ unsigned short f2bf(float f) {
  unsigned u = __builtin_bit_cast(unsigned, f);
  u += 0x7fffu + ((u >> 16) & 1u);           // round-to-nearest-even
  return (unsigned short)(u >> 16);
}
__device__ __forceinline__ float bf2f(unsigned short h) {
  return __builtin_bit_cast(float, ((unsigned)h) << 16);
}
__device__ __forceinline__ float sigmf(float x) { return 1.f / (1.f + __expf(-x)); }
__device__ __forceinline__ float hannw(int i) {
  return 0.5f * (1.f - __cosf(TWOPI * (float)i / 1024.f));
}

// ---- WMMA bf16 16x16x32 fragment helpers (ISA 7.12.2 layouts) -------------
// A (16x32): lanes 0-15 hold M=lane, K {0..7,16..23}; lanes 16-31 K {8..15,24..31}
__device__ __forceinline__ v16bf ldA(const unsigned short* row0, int lda, int lane) {
  int m  = lane & 15;
  int hi = (lane >> 4) ? 8 : 0;
  const unsigned short* p = row0 + m * lda;
  v8u lo = *(const v8u*)(p + hi);
  v8u hh = *(const v8u*)(p + 16 + hi);
  v16u u;
#pragma unroll
  for (int i = 0; i < 8; i++) { u[i] = lo[i]; u[8 + i] = hh[i]; }
  return __builtin_bit_cast(v16bf, u);
}
// B (32x16) from row-major W[N,K]: lane holds column n=(l&15), K {0..7,16..23} (l<16) or {8..15,24..31}
__device__ __forceinline__ v16bf ldB(const unsigned short* col0, int ldk, int lane) {
  int n  = lane & 15;
  int kh = (lane >> 4) * 8;
  const unsigned short* p = col0 + n * ldk;
  v8u lo = *(const v8u*)(p + kh);
  v8u hh = *(const v8u*)(p + 16 + kh);
  v16u u;
#pragma unroll
  for (int i = 0; i < 8; i++) { u[i] = lo[i]; u[8 + i] = hh[i]; }
  return __builtin_bit_cast(v16bf, u);
}
__device__ __forceinline__ v8f wmma32(v16bf a, v16bf b, v8f c) {
  return __builtin_amdgcn_wmma_f32_16x16x32_bf16(false, a, false, b, (short)0, c,
                                                 false, false);
}

// ---- shared 1024-pt complex FFT in LDS (sign=-1 fwd, +1 inv) --------------
__device__ void fft1024(float2* buf, int tid, float sign) {
#pragma unroll
  for (int s = 1; s <= 10; s++) {
    int half = 1 << (s - 1);
    for (int bb = tid; bb < 512; bb += 256) {
      int grp = bb >> (s - 1);
      int pos = bb & (half - 1);
      int i0 = (grp << s) + pos;
      int i1 = i0 + half;
      float ang = sign * TWOPI * (float)pos / (float)(half << 1);
      float c = __cosf(ang), sn = __sinf(ang);
      float2 a = buf[i0], t = buf[i1];
      float tr = c * t.x - sn * t.y;
      float ti = c * t.y + sn * t.x;
      buf[i0] = make_float2(a.x + tr, a.y + ti);
      buf[i1] = make_float2(a.x - tr, a.y - ti);
    }
    __syncthreads();
  }
}

// ---- weight prep: f32 -> bf16 with N/K zero padding -----------------------
__global__ __launch_bounds__(256) void k_prep(const float* __restrict__ src,
                                              unsigned short* __restrict__ dst,
                                              int mats, int N, int K, int dN, int Kp) {
  size_t total = (size_t)mats * dN * Kp;
  for (size_t x = (size_t)blockIdx.x * 256 + threadIdx.x; x < total;
       x += (size_t)gridDim.x * 256) {
    int kp = (int)(x % Kp);
    size_t r = x / Kp;
    int n = (int)(r % dN);
    int m = (int)(r / dN);
    float v = (n < N && kp < K) ? src[((size_t)m * N + n) * K + kp] : 0.f;
    dst[x] = f2bf(v);
  }
}

// ---- STFT: reflect pad + hann window + rFFT + |.| (bf16, K-padded) --------
__global__ __launch_bounds__(256) void k_stft(const float* __restrict__ wav,
                                              float* __restrict__ spre,
                                              float* __restrict__ spim,
                                              unsigned short* __restrict__ magp) {
  int fr = blockIdx.x, b = blockIdx.y, tid = threadIdx.x;
  __shared__ float2 buf[FFTN];
  for (int i = tid; i < FFTN; i += 256) {
    int p = fr * HOPSZ + i;
    int q = p - 512;
    if (q < 0) q = -q;                           // reflect left
    else if (q >= TLEN) q = 2 * TLEN - 2 - q;    // reflect right
    float v = wav[(size_t)b * TLEN + q] * hannw(i);
    buf[__brev((unsigned)i) >> 22] = make_float2(v, 0.f);
  }
  __syncthreads();
  fft1024(buf, tid, -1.f);
  for (int f = tid; f < FPAD; f += 256) {
    size_t om = ((size_t)b * MPAD + fr) * FPAD + f;
    if (f < FBIN) {
      size_t o = ((size_t)b * NFR + fr) * FBIN + f;
      float re = buf[f].x, im = buf[f].y;
      spre[o] = re;
      spim[o] = im;
      magp[om] = f2bf(sqrtf(re * re + im * im));
    } else {
      magp[om] = 0;                               // zero K padding
    }
  }
}

// ---- generic batched WMMA GEMM: C[b,m,n] = sum_k A[b,m,k]*W[e,n,k] + bias --
// Block tile M=128 (4 wave-rows) x N=64 (2 wave-cols); each wave owns a 32x32
// register tile (2x2 fragments -> 4 WMMAs per 8 b128 loads per K-step).
// mode 0: store C (f32).  mode 1: decoder -> Y=sigmoid(.), spec *= Y in place.
__global__ __launch_bounds__(256) void k_gemm(
    const unsigned short* __restrict__ A, int lda, size_t aStride,
    const unsigned short* __restrict__ W, size_t wStride, const int* __restrict__ S,
    const float* __restrict__ bias, int biasStride, float* __restrict__ C, int ldc,
    size_t cStride, int Kp, int Nvalid, int Mvalid, int mode,
    float* __restrict__ spre, float* __restrict__ spim) {
  int bidx = blockIdx.z;
  int wave = threadIdx.x >> 5, lane = threadIdx.x & 31;
  int m0 = blockIdx.y * 128 + (wave >> 1) * 32;
  int n0 = blockIdx.x * 64 + (wave & 1) * 32;
  int e = S ? S[bidx] : 0;
  const unsigned short* Ab = A + bidx * aStride + (size_t)m0 * lda;
  const unsigned short* Wb = W + (size_t)e * wStride + (size_t)n0 * Kp;
  v8f acc[2][2];
#pragma unroll
  for (int mi = 0; mi < 2; mi++)
#pragma unroll
    for (int ni = 0; ni < 2; ni++)
#pragma unroll
      for (int i = 0; i < 8; i++) acc[mi][ni][i] = 0.f;

  for (int k0 = 0; k0 < Kp; k0 += 32) {
    __builtin_prefetch(Wb + k0 + 64, 0, 3);       // global_prefetch_b8 (near)
    v16bf a0 = ldA(Ab + k0, lda, lane);
    v16bf a1 = ldA(Ab + (size_t)16 * lda + k0, lda, lane);
    v16bf b0 = ldB(Wb + k0, Kp, lane);
    v16bf b1 = ldB(Wb + (size_t)16 * Kp + k0, Kp, lane);
    acc[0][0] = wmma32(a0, b0, acc[0][0]);
    acc[0][1] = wmma32(a0, b1, acc[0][1]);
    acc[1][0] = wmma32(a1, b0, acc[1][0]);
    acc[1][1] = wmma32(a1, b1, acc[1][1]);
  }

#pragma unroll
  for (int mi = 0; mi < 2; mi++) {
#pragma unroll
    for (int ni = 0; ni < 2; ni++) {
      int cn = n0 + ni * 16 + (lane & 15);
      int cmb = m0 + mi * 16 + (lane >> 4) * 8;
      float bv = (bias && cn < Nvalid) ? bias[(size_t)e * biasStride + cn] : 0.f;
      if (mode == 0) {
#pragma unroll
        for (int r = 0; r < 8; r++) {
          int m = cmb + r;
          if (m < Mvalid && cn < Nvalid)
            C[bidx * cStride + (size_t)m * ldc + cn] = acc[mi][ni][r] + bv;
        }
      } else {
#pragma unroll
        for (int r = 0; r < 8; r++) {
          int m = cmb + r;
          if (m < Mvalid && cn < Nvalid) {
            float yv = sigmf(acc[mi][ni][r] + bv);
            size_t o = ((size_t)bidx * NFR + m) * FBIN + cn;
            spre[o] *= yv;
            spim[o] *= yv;
          }
        }
      }
    }
  }
}

// ---- WMMA GRU scan: up to 32 rows (samples) per block, h in LDS -----------
// Wave w owns hidden slice [32w,32w+32); its r/z/n accumulators for hidden j
// land in identical (lane,vgpr) slots -> gate math stays register-resident.
__global__ __launch_bounds__(256) void k_scan(
    const unsigned short* __restrict__ WhhAll, const float* __restrict__ bhhAll,
    const float* __restrict__ xp, unsigned short* __restrict__ hseq,
    float* __restrict__ emb, const int* __restrict__ rowmaps,
    const int* __restrict__ counts, int writeSeq, int writeEmb) {
  int e = 0, nrows = NB;
  const int* rm = nullptr;
  if (counts) {
    e = blockIdx.x;
    nrows = counts[e];
    rm = rowmaps + e * NB;
    if (nrows == 0) return;
  }
  const unsigned short* Whh = WhhAll + (size_t)e * (G3 * HID);
  const float* bhh = bhhAll + (size_t)e * G3;

  __shared__ __align__(128) unsigned short hbuf[NB][HID];
  for (int i = threadIdx.x; i < NB * HID; i += 256) ((unsigned short*)hbuf)[i] = 0;
  __syncthreads();

  int wave = threadIdx.x >> 5, lane = threadIdx.x & 31;
  int jbase = wave * 32;
  int cn = lane & 15, ch = lane >> 4;

  int smp[2][8];
#pragma unroll
  for (int mt = 0; mt < 2; mt++)
#pragma unroll
    for (int r = 0; r < 8; r++) {
      int rr = mt * 16 + ch * 8 + r;
      smp[mt][r] = (rr < nrows) ? (rm ? rm[rr] : rr) : -1;
    }

  float bh_r[2], bh_z[2], bh_n[2];
#pragma unroll
  for (int jt = 0; jt < 2; jt++) {
    int j = jbase + jt * 16 + cn;
    bh_r[jt] = bhh[j];
    bh_z[jt] = bhh[256 + j];
    bh_n[jt] = bhh[512 + j];
  }

  for (int t = 0; t < NFR; t++) {
    v8f acc[2][2][3];
#pragma unroll
    for (int mt = 0; mt < 2; mt++)
#pragma unroll
      for (int jt = 0; jt < 2; jt++)
#pragma unroll
        for (int g = 0; g < 3; g++)
#pragma unroll
          for (int i = 0; i < 8; i++) acc[mt][jt][g][i] = 0.f;

#pragma unroll
    for (int kt = 0; kt < 8; kt++) {
      int k0 = kt * 32;
      v16bf a0 = ldA(&hbuf[0][0] + k0, HID, lane);
      v16bf a1 = ldA(&hbuf[16][0] + k0, HID, lane);
#pragma unroll
      for (int jt = 0; jt < 2; jt++) {
#pragma unroll
        for (int g = 0; g < 3; g++) {
          int n0 = g * 256 + jbase + jt * 16;
          v16bf bf = ldB(Whh + (size_t)n0 * HID + k0, HID, lane);
          acc[0][jt][g] = wmma32(a0, bf, acc[0][jt][g]);
          acc[1][jt][g] = wmma32(a1, bf, acc[1][jt][g]);
        }
      }
    }
    __syncthreads();   // all A-frag reads of hbuf done before overwrite

#pragma unroll
    for (int mt = 0; mt < 2; mt++) {
#pragma unroll
      for (int r = 0; r < 8; r++) {
        int s = smp[mt][r];
        if (s < 0) continue;
        int row = mt * 16 + ch * 8 + r;
        const float* xr = xp + ((size_t)s * MPAD + t) * G3;
#pragma unroll
        for (int jt = 0; jt < 2; jt++) {
          int j = jbase + jt * 16 + cn;
          float ghr = acc[mt][jt][0][r] + bh_r[jt];
          float ghz = acc[mt][jt][1][r] + bh_z[jt];
          float ghn = acc[mt][jt][2][r] + bh_n[jt];
          float rg = sigmf(xr[j] + ghr);
          float zg = sigmf(xr[256 + j] + ghz);
          float ng = tanhf(xr[512 + j] + rg * ghn);
          float hold = bf2f(hbuf[row][j]);
          float hn = (1.f - zg) * ng + zg * hold;
          unsigned short hb = f2bf(hn);
          hbuf[row][j] = hb;
          if (writeSeq) hseq[((size_t)s * MPAD + t) * HID + j] = hb;
          if (writeEmb && t == NFR - 1) emb[s * HID + j] = hn;
        }
      }
    }
    __syncthreads();
  }
}

// ---- routing: argmin cdist + per-expert row compaction --------------------
__global__ __launch_bounds__(256) void k_route(const float* __restrict__ emb,
                                               const float* __restrict__ cm,
                                               int* __restrict__ S,
                                               int* __restrict__ counts,
                                               int* __restrict__ rowmaps) {
  int tid = threadIdx.x;
  __shared__ int sS[NB];
  if (tid < NB) {
    float best = 3.4e38f;
    int bi = 0;
    for (int n = 0; n < NEXP; n++) {
      float d = 0.f;
      for (int h = 0; h < HID; h++) {
        float df = emb[tid * HID + h] - cm[n * HID + h];
        d += df * df;
      }
      if (d < best) { best = d; bi = n; }
    }
    S[tid] = bi;
    sS[tid] = bi;
  }
  __syncthreads();
  if (tid == 0) {
    int cnt[NEXP];
    for (int x = 0; x < NEXP; x++) cnt[x] = 0;
    for (int b = 0; b < NB; b++) {
      int ee = sS[b];
      rowmaps[ee * NB + cnt[ee]] = b;
      cnt[ee]++;
    }
    for (int x = 0; x < NEXP; x++) counts[x] = cnt[x];
  }
}

// ---- iSTFT: Hermitian extend masked spec, inverse FFT, window; store the
// windowed frame back into the (now dead) spec buffers (deterministic OLA) ---
__global__ __launch_bounds__(256) void k_istft(float* __restrict__ spre,
                                               float* __restrict__ spim) {
  int fr = blockIdx.x, b = blockIdx.y, tid = threadIdx.x;
  __shared__ float2 buf[FFTN];
  size_t base = ((size_t)b * NFR + fr) * FBIN;
  for (int i = tid; i < FFTN; i += 256) {
    int src = (i <= 512) ? i : FFTN - i;
    float re = spre[base + src], im = spim[base + src];
    if (i > 512) im = -im;
    buf[__brev((unsigned)i) >> 22] = make_float2(re, im);
  }
  __syncthreads();
  fft1024(buf, tid, +1.f);
  for (int i = tid; i < FFTN; i += 256) {
    float v = buf[i].x * (1.f / 1024.f) * hannw(i);
    if (i < FBIN) spre[base + i] = v;
    else          spim[base + i - FBIN] = v;
  }
}

// ---- overlap-add + window^2 normalization + center-crop -------------------
__global__ __launch_bounds__(256) void k_ola(const float* __restrict__ fre,
                                             const float* __restrict__ fim,
                                             float* __restrict__ out) {
  int idx = blockIdx.x * 256 + threadIdx.x;
  if (idx >= NB * TLEN) return;
  int b = idx / TLEN, i = idx % TLEN;
  int p = 512 + i;
  float s = 0.f, ws2 = 0.f;
  int f0 = p >> 8;
#pragma unroll
  for (int d = 0; d < 4; d++) {
    int f = f0 - d;
    if (f >= 0 && f < NFR) {
      int ii = p - (f << 8);
      float wv = hannw(ii);
      ws2 += wv * wv;
      size_t base = ((size_t)b * NFR + f) * FBIN;
      s += (ii < FBIN) ? fre[base + ii] : fim[base + ii - FBIN];
    }
  }
  out[idx] = s / ((ws2 > 1e-11f) ? ws2 : 1.f);
}

// ---------------------------------------------------------------------------
extern "C" void kernel_launch(void* const* d_in, const int* in_sizes, int n_in,
                              void* d_out, int out_size, void* d_ws, size_t ws_size,
                              hipStream_t stream) {
  (void)in_sizes; (void)n_in; (void)out_size; (void)ws_size;
  const float* wav    = (const float*)d_in[0];
  const float* gWih0  = (const float*)d_in[1];
  const float* gWhh0  = (const float*)d_in[2];
  const float* gbih0  = (const float*)d_in[3];
  const float* gbhh0  = (const float*)d_in[4];
  const float* gWih1  = (const float*)d_in[5];
  const float* gWhh1  = (const float*)d_in[6];
  const float* gbih1  = (const float*)d_in[7];
  const float* gbhh1  = (const float*)d_in[8];
  const float* spWih0 = (const float*)d_in[9];
  const float* spWhh0 = (const float*)d_in[10];
  const float* spbih0 = (const float*)d_in[11];
  const float* spbhh0 = (const float*)d_in[12];
  const float* spWih1 = (const float*)d_in[13];
  const float* spWhh1 = (const float*)d_in[14];
  const float* spbih1 = (const float*)d_in[15];
  const float* spbhh1 = (const float*)d_in[16];
  const float* decWf  = (const float*)d_in[17];
  const float* decBf  = (const float*)d_in[18];
  const float* cmeans = (const float*)d_in[19];
  float* out = (float*)d_out;

  char* wsb = (char*)d_ws;
  size_t off = 0;
  auto take = [&](size_t n) -> char* {
    char* p = wsb + off;
    off = (off + n + 255) & ~(size_t)255;
    return p;
  };
  float* spre = (float*)take((size_t)NB * NFR * FBIN * 4);
  float* spim = (float*)take((size_t)NB * NFR * FBIN * 4);
  unsigned short* magp = (unsigned short*)take((size_t)NB * MPAD * FPAD * 2);
  float* xp = (float*)take((size_t)NB * MPAD * G3 * 4);
  unsigned short* hseq = (unsigned short*)take((size_t)NB * MPAD * HID * 2);
  float* emb = (float*)take((size_t)NB * HID * 4);
  int* S = (int*)take(NB * 4);
  int* counts = (int*)take(NEXP * 4);
  int* rowmaps = (int*)take(NEXP * NB * 4);
  unsigned short* bgW0 = (unsigned short*)take((size_t)G3 * FPAD * 2);
  unsigned short* bgU0 = (unsigned short*)take((size_t)G3 * HID * 2);
  unsigned short* bgW1 = (unsigned short*)take((size_t)G3 * HID * 2);
  unsigned short* bgU1 = (unsigned short*)take((size_t)G3 * HID * 2);
  unsigned short* bsW0 = (unsigned short*)take((size_t)NEXP * G3 * FPAD * 2);
  unsigned short* bsU0 = (unsigned short*)take((size_t)NEXP * G3 * HID * 2);
  unsigned short* bsW1 = (unsigned short*)take((size_t)NEXP * G3 * HID * 2);
  unsigned short* bsU1 = (unsigned short*)take((size_t)NEXP * G3 * HID * 2);
  unsigned short* bdW  = (unsigned short*)take((size_t)NEXP * DECNP * HID * 2);

  // weight prep (f32 -> bf16, zero K/N padding)
  k_prep<<<512, 256, 0, stream>>>(gWih0, bgW0, 1, G3, FBIN, G3, FPAD);
  k_prep<<<256, 256, 0, stream>>>(gWhh0, bgU0, 1, G3, HID, G3, HID);
  k_prep<<<256, 256, 0, stream>>>(gWih1, bgW1, 1, G3, HID, G3, HID);
  k_prep<<<256, 256, 0, stream>>>(gWhh1, bgU1, 1, G3, HID, G3, HID);
  k_prep<<<2048, 256, 0, stream>>>(spWih0, bsW0, NEXP, G3, FBIN, G3, FPAD);
  k_prep<<<1024, 256, 0, stream>>>(spWhh0, bsU0, NEXP, G3, HID, G3, HID);
  k_prep<<<1024, 256, 0, stream>>>(spWih1, bsW1, NEXP, G3, HID, G3, HID);
  k_prep<<<1024, 256, 0, stream>>>(spWhh1, bsU1, NEXP, G3, HID, G3, HID);
  k_prep<<<1024, 256, 0, stream>>>(decWf, bdW, NEXP, FBIN, HID, DECNP, HID);

  // STFT + magnitude
  k_stft<<<dim3(NFR, NB), 256, 0, stream>>>(wav, spre, spim, magp);

  dim3 gx(G3 / 64, MPAD / 128, NB);
  // gating layer 1: xp = mag @ Wih0^T + bih0 ; scan
  k_gemm<<<gx, 256, 0, stream>>>(magp, FPAD, (size_t)MPAD * FPAD, bgW0, 0, nullptr,
                                 gbih0, 0, xp, G3, (size_t)MPAD * G3, FPAD, G3, NFR,
                                 0, nullptr, nullptr);
  k_scan<<<1, 256, 0, stream>>>(bgU0, gbhh0, xp, hseq, nullptr, nullptr, nullptr, 1, 0);
  // gating layer 2
  k_gemm<<<gx, 256, 0, stream>>>(hseq, HID, (size_t)MPAD * HID, bgW1, 0, nullptr,
                                 gbih1, 0, xp, G3, (size_t)MPAD * G3, HID, G3, NFR,
                                 0, nullptr, nullptr);
  k_scan<<<1, 256, 0, stream>>>(bgU1, gbhh1, xp, hseq, emb, nullptr, nullptr, 0, 1);
  // routing
  k_route<<<1, 256, 0, stream>>>(emb, cmeans, S, counts, rowmaps);
  // specialist layer 1 (per-sample expert weights, grouped scans)
  k_gemm<<<gx, 256, 0, stream>>>(magp, FPAD, (size_t)MPAD * FPAD, bsW0,
                                 (size_t)G3 * FPAD, S, spbih0, G3, xp, G3,
                                 (size_t)MPAD * G3, FPAD, G3, NFR, 0, nullptr, nullptr);
  k_scan<<<NEXP, 256, 0, stream>>>(bsU0, spbhh0, xp, hseq, nullptr, rowmaps, counts, 1, 0);
  // specialist layer 2
  k_gemm<<<gx, 256, 0, stream>>>(hseq, HID, (size_t)MPAD * HID, bsW1,
                                 (size_t)G3 * HID, S, spbih1, G3, xp, G3,
                                 (size_t)MPAD * G3, HID, G3, NFR, 0, nullptr, nullptr);
  k_scan<<<NEXP, 256, 0, stream>>>(bsU1, spbhh1, xp, hseq, nullptr, rowmaps, counts, 1, 0);
  // decoder mask applied to spec in place
  dim3 gd(DECNP / 64, MPAD / 128, NB);
  k_gemm<<<gd, 256, 0, stream>>>(hseq, HID, (size_t)MPAD * HID, bdW,
                                 (size_t)DECNP * HID, S, decBf, FBIN, nullptr, 0, 0,
                                 HID, FBIN, NFR, 1, spre, spim);
  // inverse FFT + window (frames reuse spec buffers), then deterministic OLA
  k_istft<<<dim3(NFR, NB), 256, 0, stream>>>(spre, spim);
  k_ola<<<(NB * TLEN + 255) / 256, 256, 0, stream>>>(spre, spim, out);
}
